// TrajectoryDiscriminator_59631325938041
// MI455X (gfx1250) — matrix operand order repaired and checked by reading, not compiled, not run on hardware
//
#include <hip/hip_runtime.h>
#include <hip/hip_bf16.h>
#include <math.h>

// ---------------------------------------------------------------------------
// TrajectoryDiscriminator fused kernel for MI455X (gfx1250, wave32, WMMA)
// One workgroup (128 threads = 4 waves) per batch sample; all state in LDS.
// GEMMs -> v_wmma_f32_16x16x32_f16 ; dmat/bmat/hmat streamed one timestep
// ahead with GLOBAL_LOAD_ASYNC_TO_LDS_B128 (ASYNCcnt) double buffering.
// ---------------------------------------------------------------------------

typedef __attribute__((ext_vector_type(16))) _Float16 v16h;
typedef __attribute__((ext_vector_type(8)))  float    v8f;

#define DEV static __device__ __forceinline__

#define LOG2E 1.4426950408889634f
DEV float fexp(float x)  { return __builtin_amdgcn_exp2f(x * LOG2E); }       // v_exp_f32
DEV float sigf(float x)  { return __builtin_amdgcn_rcpf(1.f + fexp(-x)); }   // v_rcp_f32
DEV float tanhfast(float x) { return 2.f * __builtin_amdgcn_rcpf(1.f + fexp(-2.f * x)) - 1.f; }

// A-matrix fragment: 16x32 f16, lane half h holds k = 16h + s (contiguous order, same as load_bfrag).
template <typename T>
DEV v16h load_afrag(const T* A, int lda, int mt, int kb, int lane) {
  const int half = lane >> 4;
  const T* row = A + (size_t)(mt * 16 + (lane & 15)) * lda + kb;
  v16h a;
#pragma unroll
  for (int s = 0; s < 16; ++s) {
    const int k = half * 16 + s;
    a[s] = (_Float16)row[k];
  }
  return a;
}

// B-matrix fragment: 32x16 f16; lanes 0..15 rows kb+0..15, lanes 16..31 rows kb+16..31.
template <typename T>
DEV v16h load_bfrag(const T* B, int ldb, int kb, int nt, int lane) {
  const T* p = B + (size_t)(kb + ((lane >> 4) << 4)) * ldb + (nt * 16 + (lane & 15));
  v16h b;
#pragma unroll
  for (int s = 0; s < 16; ++s) b[s] = (_Float16)p[(size_t)s * ldb];
  return b;
}

// D = A[32 x K] * B[K x P] (+bias). Wave-uniform tile loop -> EXEC all ones.
template <typename TA, typename TB>
DEV void mm_wmma(const TA* A, int lda, const TB* B, int ldb, int K,
                 const float* bias, float* D, int P,
                 int wave, int nwaves, int lane) {
  const int ntiles = P >> 4;
  const int tot = 2 * ntiles;
  for (int tile = wave; tile < tot; tile += nwaves) {
    const int mt = tile / ntiles, nt = tile % ntiles;
    v8f acc = {0.f, 0.f, 0.f, 0.f, 0.f, 0.f, 0.f, 0.f};
    for (int kb = 0; kb < K; kb += 32) {
      v16h a = load_afrag<TA>(A, lda, mt, kb, lane);
      v16h b = load_bfrag<TB>(B, ldb, kb, nt, lane);
      acc = __builtin_amdgcn_wmma_f32_16x16x32_f16(
          false, a, false, b, (short)0, acc, false, false);
    }
    const int col = nt * 16 + (lane & 15);
    const int mb  = mt * 16 + ((lane >> 4) << 3);
    const float bc = bias ? bias[col] : 0.f;
#pragma unroll
    for (int r = 0; r < 8; ++r) D[(size_t)(mb + r) * P + col] = acc[r] + bc;
  }
}

// ---- LDS layout (bytes); total 152896 (<=160KB -> 2 blocks/WGP) ------------
#define OFF_HSF    0            // _Float16 [32][640] enc history         40960
#define OFF_H      40960        // float [32][32]  hidden state            4096
#define OFF_C      45056        // float [32][32]  cell state              4096
#define OFF_AG     49152        // float [32][64]  [xe|h|0]  / cat         8192
#define OFF_G      57344        // float [32][128] gates / z1 / z2        16384
#define OFF_HL     73728        // float [32][32]  lstm hidden             4096
#define OFF_HA     77824        // float [32][32]  enc2att out             4096
#define OFF_WM     81920        // float [32][32]  attention weights       4096
#define OFF_ATT    86016        // float [32][32]  att out                 4096
#define OFF_EMB    90112        // float [32][32]  spemb out               4096
#define OFF_WG     94208        // _Float16 [64][128] [w_ih;w_hh;0]       16384
#define OFF_E2A    110592       // _Float16 [32][32]                       2048
#define OFF_SPW    112640       // _Float16 [64][32]                       4096
#define OFF_A2E    116736       // _Float16 [32][32]                       2048
#define OFF_TAB    118784       // float [144]                              576
#define OFF_RS     119360       // float [32]  (reciprocal row sums)        128
#define OFF_MR     119488       // float [32]  mask row                     128
#define OFF_BG     119616       // float [128] b_ih+b_hh                    512
#define OFF_Z1H    120128       // _Float16 [32][128]                      8192
#define OFF_STAGE  128320       // float: 2 bufs x {d,b,h} x 1024         24576
#define SMEM_BYTES 152896

// Async-stage one 32x32 f32 tile (rows stride 640 floats = 2560B in global)
// into LDS at byte offset ldsOff. 256 chunks of 16B; 128 threads x 2.
DEV void issue_tile_async(const float* base, unsigned ldsOff, int tid) {
#pragma unroll
  for (int k = 0; k < 2; ++k) {
    const int c = tid + k * 128;
    const unsigned voff  = (unsigned)((c >> 3) * 2560 + (c & 7) * 16);
    const unsigned laddr = ldsOff + (unsigned)c * 16;
    asm volatile("global_load_async_to_lds_b128 %0, %1, %2"
                 :: "v"(laddr), "v"(voff), "s"(base) : "memory");
  }
}

DEV void wait_async0() {
  asm volatile("s_wait_asynccnt 0x0" ::: "memory");
}

__global__ __launch_bounds__(128) void traj_disc_kernel(
    const float* __restrict__ x,     const float* __restrict__ dmat,
    const float* __restrict__ bmat,  const float* __restrict__ hmat,
    const float* __restrict__ mask,
    const float* __restrict__ emb_w, const float* __restrict__ emb_b,
    const float* __restrict__ w_ih,  const float* __restrict__ w_hh,
    const float* __restrict__ b_ih,  const float* __restrict__ b_hh,
    const float* __restrict__ e2a_w, const float* __restrict__ e2a_b,
    const float* __restrict__ dtab,
    const float* __restrict__ sp_w,  const float* __restrict__ sp_b,
    const float* __restrict__ a2e_w, const float* __restrict__ a2e_b,
    const float* __restrict__ c1w,   const float* __restrict__ c1b,
    const float* __restrict__ c2w,   const float* __restrict__ c2b,
    const float* __restrict__ c3w,   const float* __restrict__ c3b,
    float* __restrict__ out) {
  extern __shared__ char smem[];
  _Float16* hsf  = (_Float16*)(smem + OFF_HSF);
  float*    hst  = (float*)(smem + OFF_H);
  float*    cst  = (float*)(smem + OFF_C);
  float*    Ag   = (float*)(smem + OFF_AG);   // also "cat" buffer
  float*    gbuf = (float*)(smem + OFF_G);    // gates / z1 / z2
  float*    hl   = (float*)(smem + OFF_HL);
  float*    ha   = (float*)(smem + OFF_HA);
  float*    wm   = (float*)(smem + OFF_WM);
  float*    att  = (float*)(smem + OFF_ATT);
  float*    embb = (float*)(smem + OFF_EMB);
  _Float16* Wg   = (_Float16*)(smem + OFF_WG);
  _Float16* e2ah = (_Float16*)(smem + OFF_E2A);
  _Float16* spwh = (_Float16*)(smem + OFF_SPW);
  _Float16* a2eh = (_Float16*)(smem + OFF_A2E);
  float*    tabs = (float*)(smem + OFF_TAB);
  float*    rs   = (float*)(smem + OFF_RS);
  float*    mr   = (float*)(smem + OFF_MR);
  float*    bg   = (float*)(smem + OFF_BG);
  _Float16* z1h  = (_Float16*)(smem + OFF_Z1H);
  float*    stg  = (float*)(smem + OFF_STAGE); // [2][3][1024]

  const int b    = blockIdx.x;
  const int tid  = threadIdx.x;
  const int lane = tid & 31;
  const int wave = tid >> 5;
  const int NW   = 4;

  const size_t nb = (size_t)b * 20480; // b * N*T*N
  // kick off async staging for t=0 (buffer 0) before anything else
  issue_tile_async(dmat + nb, OFF_STAGE + 0, tid);
  issue_tile_async(bmat + nb, OFF_STAGE + 4096, tid);
  issue_tile_async(hmat + nb, OFF_STAGE + 8192, tid);

  // ---- stage small weights into LDS (overlaps with async DMA) ----
  for (int i = tid; i < 64 * 128; i += 128) {
    const int r = i >> 7, cc = i & 127;
    float v = (r < 16) ? w_ih[r * 128 + cc]
                       : (r < 48 ? w_hh[(r - 16) * 128 + cc] : 0.f);
    Wg[i] = (_Float16)v;
  }
  if (tid < 128) bg[tid] = b_ih[tid] + b_hh[tid];
  for (int i = tid; i < 1024; i += 128) e2ah[i] = (_Float16)e2a_w[i];
  for (int i = tid; i < 2048; i += 128) spwh[i] = (_Float16)sp_w[i];
  for (int i = tid; i < 1024; i += 128) a2eh[i] = (_Float16)a2e_w[i];
  for (int i = tid; i < 144;  i += 128) tabs[i] = dtab[i];
  for (int i = tid; i < 1024; i += 128) { hst[i] = 0.f; cst[i] = 0.f; }
  __syncthreads();

  // ---- recurrent scan over T=20 ----
  for (int t = 0; t < 20; ++t) {
    const int sel = t & 1;
    const float* dS = stg + sel * 3072;        // [1024] d tile
    const float* bS = dS + 1024;               // [1024] bearing tile
    const float* hS = dS + 2048;               // [1024] heading tile

    // make this wave's async loads for step t complete
    wait_async0();

    // (1) Ag = [xe | h | 0]
    for (int i = tid; i < 32 * 64; i += 128) {
      const int n = i >> 6, cc = i & 63;
      float v;
      if (cc < 16) {
        const float* xp = x + (((size_t)b * 32 + n) * 20 + t) * 2;
        v = xp[0] * emb_w[cc] + xp[1] * emb_w[16 + cc] + emb_b[cc];
      } else if (cc < 48) {
        v = hst[n * 32 + (cc - 16)];
      } else {
        v = 0.f;
      }
      Ag[i] = v;
    }
    if (tid < 32) mr[tid] = mask[((size_t)b * 32 + tid) * 20 + t];
    __syncthreads();   // all waves' async data now visible too

    // issue async staging for t+1 into the other buffer (overlaps the step)
    if (t < 19) {
      const unsigned soff = OFF_STAGE + (unsigned)((t + 1) & 1) * 12288;
      const size_t goff = nb + (size_t)(t + 1) * 32;
      issue_tile_async(dmat + goff, soff, tid);
      issue_tile_async(bmat + goff, soff + 4096, tid);
      issue_tile_async(hmat + goff, soff + 8192, tid);
    }

    // (2) gates[32,128] = Ag[32,64] @ Wg[64,128] + (b_ih+b_hh)
    mm_wmma<float, _Float16>(Ag, 64, Wg, 128, 64, bg, gbuf, 128, wave, NW, lane);
    __syncthreads();

    // (3) LSTM pointwise (fast hw exp/rcp)
    for (int i = tid; i < 1024; i += 128) {
      const int n = i >> 5, k = i & 31;
      const float* g = gbuf + n * 128;
      const float gi = sigf(g[k]),           gf = sigf(g[32 + k]);
      const float gg = tanhfast(g[64 + k]),  go = sigf(g[96 + k]);
      const float cn = gf * cst[i] + gi * gg;
      cst[i] = cn;
      hl[i]  = go * tanhfast(cn);
    }
    __syncthreads();

    // (4) ha = h_lstm @ enc2att + b
    mm_wmma<float, _Float16>(hl, 32, e2ah, 32, 32, e2a_b, ha, 32, wave, NW, lane);

    // (5) attention weights from LDS-staged tiles
    for (int i = tid; i < 1024; i += 128) {
      const int ii = i >> 5, jj = i & 31;
      const float d = dS[i], bv = bS[i], hv = hS[i];
      int ib = (int)(bv * (1.f / 30.f)); ib = ib < 0 ? 0 : (ib > 11 ? 11 : ib);
      int ih = (int)(hv * (1.f / 30.f)); ih = ih < 0 ? 0 : (ih > 11 ? 11 : ih);
      float wv = tabs[ib * 12 + ih] - d;
      wv = wv > 0.f ? wv : 0.f;
      wm[i] = wv * mr[ii] * mr[jj];
    }
    __syncthreads();

    // (6) reciprocal row sums, (7) normalize
    if (tid < 32) {
      float s = 1e-8f;
      for (int j = 0; j < 32; ++j) s += wm[tid * 32 + j];
      rs[tid] = __builtin_amdgcn_rcpf(s);
    }
    __syncthreads();
    for (int i = tid; i < 1024; i += 128) wm[i] *= rs[i >> 5];
    __syncthreads();

    // (8) att = w @ ha
    mm_wmma<float, float>(wm, 32, ha, 32, 32, nullptr, att, 32, wave, NW, lane);
    __syncthreads();

    // (9) cat = [ha | att]  (reuses Ag)
    for (int i = tid; i < 2048; i += 128) {
      const int n = i >> 6, cc = i & 63;
      Ag[i] = (cc < 32) ? ha[n * 32 + cc] : att[n * 32 + cc - 32];
    }
    __syncthreads();

    // (10) emb = tanh(cat @ spemb + b)
    mm_wmma<float, _Float16>(Ag, 64, spwh, 32, 64, sp_b, embb, 32, wave, NW, lane);
    __syncthreads();
    for (int i = tid; i < 1024; i += 128) embb[i] = tanhfast(embb[i]);
    __syncthreads();

    // (12) h_new = emb @ att2enc + b
    mm_wmma<float, _Float16>(embb, 32, a2eh, 32, 32, a2e_b, hst, 32, wave, NW, lane);
    __syncthreads();

    // (13) stash in enc layout [n][t*32+k] as f16
    for (int i = tid; i < 1024; i += 128) {
      const int n = i >> 5, k = i & 31;
      hsf[n * 640 + t * 32 + k] = (_Float16)hst[i];
    }
    __syncthreads();
  }

  // ---- classifier MLP ----
  // z1 = enc[32,640] @ cls_w1[640,128] + b1  (cls_w1 L2-resident, shared by all blocks)
  mm_wmma<_Float16, float>(hsf, 640, c1w, 128, 640, c1b, gbuf, 128, wave, NW, lane);
  __syncthreads();
  for (int i = tid; i < 4096; i += 128) {
    const float v = gbuf[i];
    z1h[i] = (_Float16)(v > 0.f ? v : 0.2f * v);
  }
  __syncthreads();
  // z2 = z1 @ cls_w2 + b2   (writes back into gbuf; A is z1h)
  mm_wmma<_Float16, float>(z1h, 128, c2w, 128, 128, c2b, gbuf, 128, wave, NW, lane);
  __syncthreads();
  // out = sigmoid(lrelu(z2) @ cls_w3 + b3)
  if (tid < 32) {
    float s = c3b[0];
    for (int k = 0; k < 128; ++k) {
      float v = gbuf[tid * 128 + k];
      v = v > 0.f ? v : 0.2f * v;
      s += v * c3w[k];
    }
    out[(size_t)b * 32 + tid] = sigf(s);
  }
}

extern "C" void kernel_launch(void* const* d_in, const int* in_sizes, int n_in,
                              void* d_out, int out_size, void* d_ws, size_t ws_size,
                              hipStream_t stream) {
  (void)in_sizes; (void)n_in; (void)out_size; (void)d_ws; (void)ws_size;
  const float* x     = (const float*)d_in[0];
  const float* dmat  = (const float*)d_in[1];
  const float* bmat  = (const float*)d_in[2];
  const float* hmat  = (const float*)d_in[3];
  const float* mask  = (const float*)d_in[4];
  const float* emb_w = (const float*)d_in[5];
  const float* emb_b = (const float*)d_in[6];
  const float* w_ih  = (const float*)d_in[7];
  const float* w_hh  = (const float*)d_in[8];
  const float* b_ih  = (const float*)d_in[9];
  const float* b_hh  = (const float*)d_in[10];
  const float* e2a_w = (const float*)d_in[11];
  const float* e2a_b = (const float*)d_in[12];
  const float* dtab  = (const float*)d_in[13];
  const float* sp_w  = (const float*)d_in[14];
  const float* sp_b  = (const float*)d_in[15];
  const float* a2e_w = (const float*)d_in[16];
  const float* a2e_b = (const float*)d_in[17];
  const float* c1w   = (const float*)d_in[18];
  const float* c1b   = (const float*)d_in[19];
  const float* c2w   = (const float*)d_in[20];
  const float* c2b   = (const float*)d_in[21];
  const float* c3w   = (const float*)d_in[22];
  const float* c3b   = (const float*)d_in[23];
  float* out = (float*)d_out;

  traj_disc_kernel<<<512, 128, SMEM_BYTES, stream>>>(
      x, dmat, bmat, hmat, mask, emb_w, emb_b, w_ih, w_hh, b_ih, b_hh,
      e2a_w, e2a_b, dtab, sp_w, sp_b, a2e_w, a2e_b,
      c1w, c1b, c2w, c2b, c3w, c3b, out);
}
